// GAT_value_91311004713447
// MI455X (gfx1250) — compile-verified, matrix operand order checked by reading, not run on hardware
//
#include <hip/hip_runtime.h>
#include <hip/hip_fp16.h>

// ---------------- problem constants (match reference) ----------------
#define N_NODES 100000
#define N_EDGES 1600000
#define NLAYERS 3
#define NHEADS  4
#define DHEAD   32
#define NFEAT   128          // NHEADS*DHEAD
#define NEG_SLOPE 0.2f
#define BN_EPS    1e-5f

typedef __attribute__((ext_vector_type(16))) _Float16 v16h;
typedef __attribute__((ext_vector_type(8)))  _Float16 v8h;
typedef __attribute__((ext_vector_type(8)))  float    v8f;

__device__ __forceinline__ float lrelu(float x) { return x > 0.f ? x : NEG_SLOPE * x; }

// order-preserving float<->uint encoding so we can use native u32 atomic max.
// memset(0) initializes to "below every encodable float".
__device__ __forceinline__ unsigned enc_f32(float f) {
  unsigned u = __float_as_uint(f);
  return (u & 0x80000000u) ? ~u : (u | 0x80000000u);
}
__device__ __forceinline__ float dec_f32(unsigned u) {
  return __uint_as_float((u & 0x80000000u) ? (u ^ 0x80000000u) : ~u);
}

// ---------------- f16 conversion kernels ----------------
__global__ __launch_bounds__(256) void cvt_h16(const float* __restrict__ hin,
                                               _Float16* __restrict__ h16) {
  int t = blockIdx.x * blockDim.x + threadIdx.x;          // N*128 exact
  h16[t] = (_Float16)hin[t];
}

// Wt16[n][k] = W[k][n], converted to f16.  W is [128][128] row-major (k-major).
__global__ __launch_bounds__(256) void cvt_wT16(const float* __restrict__ W,
                                                _Float16* __restrict__ Wt16) {
  int t = blockIdx.x * blockDim.x + threadIdx.x;          // 16384 exact
  int n = t >> 7, k = t & 127;
  Wt16[t] = (_Float16)W[k * NFEAT + n];
}

// ---------------- WMMA GEMM: hproj[N,128] = h16[N,128] x W[128,128] ----------------
// block = 256 thr = 8 waves; block computes rows [16*bx,16*bx+16), wave w -> cols [16w,16w+16)
__global__ __launch_bounds__(256) void gemm_h_wmma(const _Float16* __restrict__ A16,
                                                   const _Float16* __restrict__ Bt16,
                                                   float* __restrict__ Cout) {
  const int wave = threadIdx.x >> 5;
  const int lane = threadIdx.x & 31;
  const int l16  = lane & 15;
  const int hi8  = (lane >> 4) ? 8 : 0;   // lane halves own K/row halves per ISA layout
  const int m    = blockIdx.x * 16 + l16;
  const int n    = wave * 16 + l16;

  const _Float16* __restrict__ arow = A16  + (size_t)m * NFEAT;
  const _Float16* __restrict__ brow = Bt16 + (size_t)n * NFEAT;

  v8f acc = {};
#pragma unroll
  for (int kk = 0; kk < NFEAT; kk += 32) {
    const int klo = kk + hi8;
    const int khi = kk + 16 + hi8;
    v8h alo = *(const v8h*)(arow + klo);
    v8h ahi = *(const v8h*)(arow + khi);
    v8h blo = *(const v8h*)(brow + klo);
    v8h bhi = *(const v8h*)(brow + khi);
    v16h a = __builtin_shufflevector(alo, ahi, 0,1,2,3,4,5,6,7,8,9,10,11,12,13,14,15);
    v16h b = __builtin_shufflevector(blo, bhi, 0,1,2,3,4,5,6,7,8,9,10,11,12,13,14,15);
    acc = __builtin_amdgcn_wmma_f32_16x16x32_f16(false, a, false, b, (short)0, acc,
                                                 false, false);
  }
  // C layout: VGPR r, lanes 0-15 -> M=r, lanes 16-31 -> M=8+r; lane%16 = N
  float* crow = Cout + (size_t)(blockIdx.x * 16 + hi8) * NFEAT + wave * 16 + l16;
#pragma unroll
  for (int r = 0; r < 8; ++r) crow[(size_t)r * NFEAT] = acc[r];
}

// ---------------- per-node attention scores: s,d [N,H] ----------------
__global__ __launch_bounds__(256) void node_scores(const float* __restrict__ hproj,
                                                   const float* __restrict__ a_s,
                                                   const float* __restrict__ a_d,
                                                   float* __restrict__ s,
                                                   float* __restrict__ d) {
  int t = blockIdx.x * blockDim.x + threadIdx.x;          // N*H
  if (t >= N_NODES * NHEADS) return;
  int n = t >> 2, h = t & 3;
  const float* hp = hproj + (size_t)n * NFEAT + h * DHEAD;
  const float* as = a_s + h * DHEAD;
  const float* ad = a_d + h * DHEAD;
  float ss = 0.f, dd = 0.f;
#pragma unroll
  for (int i = 0; i < DHEAD; ++i) { ss += hp[i] * as[i]; dd += hp[i] * ad[i]; }
  s[t] = ss; d[t] = dd;
}

// ---------------- edge passes (H=4 layers) ----------------
__global__ __launch_bounds__(256) void edge_max(const int* __restrict__ src,
                                                const int* __restrict__ dst,
                                                const float* __restrict__ s,
                                                const float* __restrict__ d,
                                                unsigned* __restrict__ menc) {
  int t = blockIdx.x * blockDim.x + threadIdx.x;          // E*H exact
  int e = t >> 2, h = t & 3;
  int u = src[e], v = dst[e];
  float sc = lrelu(s[u * 4 + h] + d[v * 4 + h]);
  atomicMax(&menc[v * 4 + h], enc_f32(sc));
}

__global__ __launch_bounds__(256) void edge_sum(const int* __restrict__ src,
                                                const int* __restrict__ dst,
                                                const float* __restrict__ s,
                                                const float* __restrict__ d,
                                                const unsigned* __restrict__ menc,
                                                float* __restrict__ z) {
  int t = blockIdx.x * blockDim.x + threadIdx.x;          // E*H exact
  int e = t >> 2, h = t & 3;
  int u = src[e], v = dst[e];
  float sc = lrelu(s[u * 4 + h] + d[v * 4 + h]);
  float p = __expf(sc - dec_f32(menc[v * 4 + h]));
  atomicAdd(&z[v * 4 + h], p);
}

// one wave per edge: 32 lanes x 4 feats = 128; src/dst are wave-uniform -> scalar loads; m/z vary with head h = q>>3
__global__ __launch_bounds__(256) void edge_scatter(const int* __restrict__ src,
                                                    const int* __restrict__ dst,
                                                    const float* __restrict__ s,
                                                    const float* __restrict__ d,
                                                    const unsigned* __restrict__ menc,
                                                    const float* __restrict__ z,
                                                    const float* __restrict__ hproj,
                                                    float* __restrict__ agg) {
  int t = blockIdx.x * blockDim.x + threadIdx.x;          // E*32 exact
  int e = t >> 5, q = t & 31;
  int f = q << 2, h = q >> 3;
  int u = src[e], v = dst[e];
  float sc = lrelu(s[u * 4 + h] + d[v * 4 + h]);
  float alpha = __expf(sc - dec_f32(menc[v * 4 + h])) / (z[v * 4 + h] + 1e-16f);
  const float4 hv = *(const float4*)(hproj + (size_t)u * NFEAT + f);
  float* o = agg + (size_t)v * NFEAT + f;
  atomicAdd(o + 0, hv.x * alpha);
  atomicAdd(o + 1, hv.y * alpha);
  atomicAdd(o + 2, hv.z * alpha);
  atomicAdd(o + 3, hv.w * alpha);
}

// ---------------- epilogue: bias + BN + residual + ELU ----------------
__global__ __launch_bounds__(256) void post_layer(const float* __restrict__ agg,
                                                  const float* __restrict__ hin,
                                                  const float* __restrict__ bias,
                                                  const float* __restrict__ gmm,
                                                  const float* __restrict__ bet,
                                                  const float* __restrict__ mea,
                                                  const float* __restrict__ var,
                                                  float* __restrict__ hout,
                                                  int use_res) {
  int t = blockIdx.x * blockDim.x + threadIdx.x;          // N*128 exact
  int f = t & 127;
  float v = agg[t] + bias[f];
  v = gmm[f] * (v - mea[f]) * rsqrtf(var[f] + BN_EPS) + bet[f];
  if (use_res) v += hin[t];
  hout[t] = v > 0.f ? v : (__expf(v) - 1.f);
}

// ---------------- output layer (H=1, DH=1) ----------------
// one wave per node: hout[n] = dot(h[n,:], W_out)
__global__ __launch_bounds__(256) void node_proj(const float* __restrict__ h,
                                                 const float* __restrict__ Wout,
                                                 float* __restrict__ hout) {
  int node = blockIdx.x * 8 + (threadIdx.x >> 5);         // N/8 blocks exact
  int lane = threadIdx.x & 31;
  const float4 hv = *(const float4*)(h + (size_t)node * NFEAT + lane * 4);
  const float4 wv = *(const float4*)(Wout + lane * 4);
  float sum = hv.x * wv.x + hv.y * wv.y + hv.z * wv.z + hv.w * wv.w;
#pragma unroll
  for (int o = 16; o > 0; o >>= 1) sum += __shfl_xor(sum, o, 32);
  if (lane == 0) hout[node] = sum;
}

__global__ __launch_bounds__(256) void edge_max1(const int* __restrict__ src,
                                                 const int* __restrict__ dst,
                                                 const float* __restrict__ hout,
                                                 const float* __restrict__ as,
                                                 const float* __restrict__ ad,
                                                 unsigned* __restrict__ menc) {
  int e = blockIdx.x * blockDim.x + threadIdx.x;          // E exact
  int u = src[e], v = dst[e];
  float sc = lrelu(hout[u] * as[0] + hout[v] * ad[0]);
  atomicMax(&menc[v], enc_f32(sc));
}

__global__ __launch_bounds__(256) void edge_sum1(const int* __restrict__ src,
                                                 const int* __restrict__ dst,
                                                 const float* __restrict__ hout,
                                                 const float* __restrict__ as,
                                                 const float* __restrict__ ad,
                                                 const unsigned* __restrict__ menc,
                                                 float* __restrict__ z) {
  int e = blockIdx.x * blockDim.x + threadIdx.x;          // E exact
  int u = src[e], v = dst[e];
  float sc = lrelu(hout[u] * as[0] + hout[v] * ad[0]);
  atomicAdd(&z[v], __expf(sc - dec_f32(menc[v])));
}

__global__ __launch_bounds__(256) void edge_scatter1(const int* __restrict__ src,
                                                     const int* __restrict__ dst,
                                                     const float* __restrict__ hout,
                                                     const float* __restrict__ as,
                                                     const float* __restrict__ ad,
                                                     const unsigned* __restrict__ menc,
                                                     const float* __restrict__ z,
                                                     float* __restrict__ agg) {
  int e = blockIdx.x * blockDim.x + threadIdx.x;          // E exact
  int u = src[e], v = dst[e];
  float sc = lrelu(hout[u] * as[0] + hout[v] * ad[0]);
  float alpha = __expf(sc - dec_f32(menc[v])) / (z[v] + 1e-16f);
  atomicAdd(&agg[v], alpha * hout[u]);
}

__global__ __launch_bounds__(256) void final_out(const float* __restrict__ agg,
                                                 const float* __restrict__ b_out,
                                                 float* __restrict__ out) {
  int n = blockIdx.x * blockDim.x + threadIdx.x;
  if (n >= N_NODES) return;
  out[n] = agg[n] + b_out[0];
}

// ---------------- host side ----------------
extern "C" void kernel_launch(void* const* d_in, const int* in_sizes, int n_in,
                              void* d_out, int out_size, void* d_ws, size_t ws_size,
                              hipStream_t stream) {
  const float* x        = (const float*)d_in[0];
  const int*   ei       = (const int*)  d_in[1];
  const float* W        = (const float*)d_in[2];   // [3,128,128]
  const float* att_src  = (const float*)d_in[3];   // [3,4,32]
  const float* att_dst  = (const float*)d_in[4];
  const float* bias     = (const float*)d_in[5];   // [3,128]
  const float* bn_gamma = (const float*)d_in[6];
  const float* bn_beta  = (const float*)d_in[7];
  const float* bn_mean  = (const float*)d_in[8];
  const float* bn_var   = (const float*)d_in[9];
  const float* W_out    = (const float*)d_in[10];  // [128,1]
  const float* as_out   = (const float*)d_in[11];
  const float* ad_out   = (const float*)d_in[12];
  const float* b_out    = (const float*)d_in[13];

  const int* src = ei;
  const int* dst = ei + N_EDGES;

  // workspace carve-up
  char* ws = (char*)d_ws;
  size_t off = 0;
  auto carve = [&](size_t bytes) -> void* {
    void* p = ws + off;
    off += (bytes + 255) & ~(size_t)255;
    return p;
  };
  const size_t NB = (size_t)N_NODES * NFEAT * sizeof(float);
  float*    hA    = (float*)   carve(NB);                        // current h
  float*    hproj = (float*)   carve(NB);                        // projected h
  float*    agg   = (float*)   carve(NB);                        // aggregation buffer
  _Float16* h16   = (_Float16*)carve((size_t)N_NODES * NFEAT * 2);
  _Float16* w16t  = (_Float16*)carve((size_t)NFEAT * NFEAT * 2); // transposed f16 W
  float*    sbuf  = (float*)   carve((size_t)N_NODES * NHEADS * sizeof(float));
  float*    dbuf  = (float*)   carve((size_t)N_NODES * NHEADS * sizeof(float));
  unsigned* menc  = (unsigned*)carve((size_t)N_NODES * NHEADS * sizeof(unsigned));
  float*    zbuf  = (float*)   carve((size_t)N_NODES * NHEADS * sizeof(float));

  const int B = 256;
  const int gNF   = (N_NODES * NFEAT) / B;   // 50000
  const int gW    = (NFEAT * NFEAT) / B;     // 64
  const int gGemm = N_NODES / 16;            // 6250
  const int gNH   = (N_NODES * NHEADS + B - 1) / B;
  const int gEH   = (N_EDGES * NHEADS) / B;  // 25000
  const int gE32  = (N_EDGES * 32) / B;      // 200000
  const int gE    = N_EDGES / B;             // 6250

  for (int l = 0; l < NLAYERS; ++l) {
    const float* hin = (l == 0) ? x : hA;
    cvt_h16 <<<gNF, B, 0, stream>>>(hin, h16);
    cvt_wT16<<<gW,  B, 0, stream>>>(W + (size_t)l * NFEAT * NFEAT, w16t);
    gemm_h_wmma<<<gGemm, B, 0, stream>>>(h16, w16t, hproj);
    node_scores<<<gNH, B, 0, stream>>>(hproj,
                                       att_src + l * NHEADS * DHEAD,
                                       att_dst + l * NHEADS * DHEAD, sbuf, dbuf);
    hipMemsetAsync(menc, 0, (size_t)N_NODES * NHEADS * sizeof(unsigned), stream);
    hipMemsetAsync(zbuf, 0, (size_t)N_NODES * NHEADS * sizeof(float), stream);
    hipMemsetAsync(agg,  0, NB, stream);
    edge_max    <<<gEH,  B, 0, stream>>>(src, dst, sbuf, dbuf, menc);
    edge_sum    <<<gEH,  B, 0, stream>>>(src, dst, sbuf, dbuf, menc, zbuf);
    edge_scatter<<<gE32, B, 0, stream>>>(src, dst, sbuf, dbuf, menc, zbuf, hproj, agg);
    post_layer  <<<gNF,  B, 0, stream>>>(agg, hin,
                                         bias    + l * NFEAT, bn_gamma + l * NFEAT,
                                         bn_beta + l * NFEAT, bn_mean  + l * NFEAT,
                                         bn_var  + l * NFEAT, hA, l > 0 ? 1 : 0);
  }

  // output GAT layer (H=1, DH=1): reuse small buffers
  float*    hsc  = sbuf;          // [N] projected scalar
  unsigned* m1   = menc;          // [N]
  float*    z1   = zbuf;          // [N]
  float*    agg1 = dbuf;          // [N]
  node_proj<<<N_NODES / 8, B, 0, stream>>>(hA, W_out, hsc);
  hipMemsetAsync(m1,   0, (size_t)N_NODES * sizeof(unsigned), stream);
  hipMemsetAsync(z1,   0, (size_t)N_NODES * sizeof(float), stream);
  hipMemsetAsync(agg1, 0, (size_t)N_NODES * sizeof(float), stream);
  edge_max1    <<<gE, B, 0, stream>>>(src, dst, hsc, as_out, ad_out, m1);
  edge_sum1    <<<gE, B, 0, stream>>>(src, dst, hsc, as_out, ad_out, m1, z1);
  edge_scatter1<<<gE, B, 0, stream>>>(src, dst, hsc, as_out, ad_out, m1, z1, agg1);
  final_out<<<(N_NODES + B - 1) / B, B, 0, stream>>>(agg1, b_out, (float*)d_out);
}
